// message_pass_22548578304895
// MI455X (gfx1250) — matrix operand order repaired and checked by reading, not compiled, not run on hardware
//
#include <hip/hip_runtime.h>

typedef __bf16 bf16;
typedef __attribute__((ext_vector_type(16))) __bf16 v16bf;
typedef __attribute__((ext_vector_type(8)))  float  v8f;

#define E_TOT  500000
#define C_DIM  64
#define CE_DIM 16
#define DIN    144   // 2*C + CE
#define HID    128
#define KP1    160   // DIN padded to multiple of 32
#define MT1    8     // HID/16  (layer-1 computed transposed: M = HID)
#define KT1    5     // KP1/32
#define NT2    4     // C_DIM/16
#define KT2    4     // HID/32

__device__ __forceinline__ v8f wmma_bf16(v16bf a, v16bf b, v8f c) {
  // D = A(16x32 bf16) * B(32x16 bf16) + C(16x16 f32)
  return __builtin_amdgcn_wmma_f32_16x16x32_bf16(
      /*neg_a=*/false, a, /*neg_b=*/false, b,
      /*c_mod=*/(short)0, c, /*reuse_a=*/false, /*reuse_b=*/false);
}

__global__ __launch_bounds__(256) void fused_edge_mlp(
    const float* __restrict__ x_i, const float* __restrict__ x_j,
    const float* __restrict__ ea,
    const float* __restrict__ W1, const float* __restrict__ b1,
    const float* __restrict__ W2, const float* __restrict__ b2,
    const float* __restrict__ alphap, float* __restrict__ out)
{
  // W1^T pre-swizzled into per-lane A-fragment layout:
  //   frag(mt,kt): lane L, elem e (0..15):
  //     K = 32*kt + 16*(L/16) + e                  (input dim, zero-padded past 144)
  //     M = 16*mt + L%16                            (hidden dim)
  __shared__ __align__(32) bf16 sW1t[MT1 * KT1 * 32 * 16];  // 40960 B
  // W2 pre-swizzled into per-lane B-fragment layout:
  //   frag(nt,kt): lane L, elem e: K = 32*kt + 16*(e>>3) + 8*(L/16) + (e&7) ; N = 16*nt + L%16
  __shared__ __align__(32) bf16 sW2[NT2 * KT2 * 32 * 16];   // 16384 B
  __shared__ __align__(32) float sB1[HID];
  __shared__ __align__(32) float sB2[C_DIM];

  const int tid = threadIdx.x;

  // ---- one-time per-block weight conversion + swizzle (f32 -> bf16) ----
  for (int idx = tid; idx < MT1 * KT1 * 32 * 16; idx += 256) {
    int e    = idx & 15;
    int lane = (idx >> 4) & 31;
    int kt   = (idx >> 9) % KT1;
    int mt   = idx / (512 * KT1);
    int k = kt * 32 + (lane >> 4) * 16 + e;
    int m = mt * 16 + (lane & 15);
    float v = (k < DIN) ? W1[k * HID + m] : 0.0f;   // zero-pad K 144..159
    sW1t[idx] = (bf16)v;
  }
  for (int idx = tid; idx < NT2 * KT2 * 32 * 16; idx += 256) {
    int e    = idx & 15;
    int lane = (idx >> 4) & 31;
    int kt   = (idx >> 9) % KT2;
    int nt   = idx / (512 * KT2);
    int k = kt * 32 + (e >> 3) * 16 + (lane >> 4) * 8 + (e & 7);
    int n = nt * 16 + (lane & 15);
    sW2[idx] = (bf16)W2[k * C_DIM + n];
  }
  if (tid < HID)   sB1[tid] = b1[tid];
  if (tid < C_DIM) sB2[tid] = b2[tid];
  __syncthreads();

  const float alpha = alphap[0];
  const int wave = tid >> 5;
  const int lane = tid & 31;
  const int lo   = lane & 15;   // edge index within tile (both layers)
  const int hi   = lane >> 4;   // half-wave selector

  const int numTiles = E_TOT / 16;  // 31250
  for (int tile = blockIdx.x * 8 + wave; tile < numTiles; tile += gridDim.x * 8) {
    // Defeat LICM of the (loop-invariant-address) weight-fragment LDS loads:
    // without this, the compiler hoists 56 fragments (448 VGPRs) into registers
    // and spills them to scratch. Reloading from LDS each iteration is cheap.
    asm volatile("" ::: "memory");

    const int rowBase = tile * 16;
    const int r = rowBase + lo;

    // ---- layer-1 B fragments = concat^T: lane holds 16 contiguous f32 of
    //      its edge's concat row, K = 32*kt + 16*hi .. +15  (f32 -> bf16) ----
    v16bf bfrag[KT1];
    #pragma unroll
    for (int kt = 0; kt < KT1; ++kt) {
      const int k0 = kt * 32 + hi * 16;   // 16-aligned; single source each
      float vals[16];
      const float* p = nullptr;
      if (k0 < C_DIM)          p = x_i + r * C_DIM + k0;
      else if (k0 < 2 * C_DIM) p = x_j + r * C_DIM + (k0 - C_DIM);
      else if (k0 < DIN)       p = ea  + r * CE_DIM + (k0 - 2 * C_DIM);
      if (p) {
        #pragma unroll
        for (int q = 0; q < 4; ++q) {
          float4 u = *(const float4*)(p + 4 * q);
          vals[4*q+0] = u.x; vals[4*q+1] = u.y; vals[4*q+2] = u.z; vals[4*q+3] = u.w;
        }
      } else {
        #pragma unroll
        for (int i = 0; i < 16; ++i) vals[i] = 0.0f;
      }
      #pragma unroll
      for (int i = 0; i < 16; ++i) bfrag[kt][i] = (bf16)vals[i];
    }

    // ---- layer 1 (transposed): h^T = W1^T @ concat^T.
    //      acc tiles for mt = 2*kt2, 2*kt2+1 ARE the layer-2 A fragment kt2. ----
    v16bf a2[KT2];
    #pragma unroll
    for (int kt2 = 0; kt2 < KT2; ++kt2) {
      #pragma unroll
      for (int sub = 0; sub < 2; ++sub) {
        const int mt = kt2 * 2 + sub;
        v8f acc = {};
        #pragma unroll
        for (int kt = 0; kt < KT1; ++kt) {
          const v16bf w1f = *(const v16bf*)&sW1t[((mt * KT1 + kt) * 32 + lane) * 16];
          acc = wmma_bf16(w1f, bfrag[kt], acc);
        }
        const v8f bv = *(const v8f*)&sB1[mt * 16 + hi * 8];
        #pragma unroll
        for (int v = 0; v < 8; ++v) {
          float x = acc[v] + bv[v];
          x = (x >= 0.0f) ? x : x * alpha;
          a2[kt2][sub * 8 + v] = (bf16)x;   // in-register layer-2 A fragment
        }
      }
    }

    // ---- layer 2 (normal): out = h @ W2, coalesced f32 stores ----
    #pragma unroll
    for (int nt = 0; nt < NT2; ++nt) {
      const float bias = sB2[nt * 16 + lo];
      v8f acc;
      #pragma unroll
      for (int v = 0; v < 8; ++v) acc[v] = bias;
      #pragma unroll
      for (int kt = 0; kt < KT2; ++kt) {
        const v16bf w2f = *(const v16bf*)&sW2[((nt * KT2 + kt) * 32 + lane) * 16];
        acc = wmma_bf16(a2[kt], w2f, acc);
      }
      #pragma unroll
      for (int v = 0; v < 8; ++v) {
        float x = acc[v];
        x = (x >= 0.0f) ? x : x * alpha;
        out[(rowBase + hi * 8 + v) * C_DIM + nt * 16 + lo] = x;
      }
    }
  }
}

extern "C" void kernel_launch(void* const* d_in, const int* in_sizes, int n_in,
                              void* d_out, int out_size, void* d_ws, size_t ws_size,
                              hipStream_t stream) {
  (void)in_sizes; (void)n_in; (void)out_size; (void)d_ws; (void)ws_size;
  const float* x_i   = (const float*)d_in[0];
  const float* x_j   = (const float*)d_in[1];
  const float* ea    = (const float*)d_in[2];
  const float* W1    = (const float*)d_in[3];
  const float* b1    = (const float*)d_in[4];
  const float* W2    = (const float*)d_in[5];
  const float* b2    = (const float*)d_in[6];
  const float* alpha = (const float*)d_in[7];
  float* out = (float*)d_out;

  dim3 grid(512), block(256);
  fused_edge_mlp<<<grid, block, 0, stream>>>(x_i, x_j, ea, W1, b1, W2, b2, alpha, out);
}
